// OPTBlock_17291538334268
// MI455X (gfx1250) — compile-verified, matrix operand order checked by reading, not run on hardware
//
#include <hip/hip_runtime.h>

#define N_PTS   32768
#define C_DIM   256
#define H_HEADS 8
#define D_HEAD  32
#define WIN_    256
#define STRIDE_ 128
#define M_WIN   256
#define C3      768
#define C2      512
#define EPSF    1e-5f

typedef __attribute__((ext_vector_type(16))) __bf16 v16bf;
typedef __attribute__((ext_vector_type(8)))  __bf16 v8bf;
typedef __attribute__((ext_vector_type(8)))  float  v8f;
typedef int v4i __attribute__((vector_size(16)));

#define GLOBAL_AS __attribute__((address_space(1)))
#define LDS_AS    __attribute__((address_space(3)))

#if __has_builtin(__builtin_amdgcn_global_load_async_to_lds_b128)
#define HAS_ASYNC_LDS 1
#endif

// ---------------------------------------------------------------------------
// WMMA helpers (CDNA5 16x16x32 bf16 -> f32)
// A-fragment layout (16-bit, 16x32): lanes 0-15 hold K 0-7 (h0..7) and K 16-23
// (h8..15); lanes 16-31 hold K 8-15 and K 24-31.  => two contiguous 16B loads
// at +koff and +koff+16 where koff = (lane>=16) ? 8 : 0.
// B-fragment is the same pattern on Bt[N][K] (column-major B).
// ---------------------------------------------------------------------------
static __device__ __forceinline__ v16bf load_frag16(const __bf16* p) {
  v8bf lo = *(const v8bf*)(p);
  v8bf hi = *(const v8bf*)(p + 16);
  v16bf f;
#pragma unroll
  for (int i = 0; i < 8; ++i) { f[i] = lo[i]; f[i + 8] = hi[i]; }
  return f;
}

static __device__ __forceinline__ v8f wmma_bf16(v16bf a, v16bf b, v8f c) {
  return __builtin_amdgcn_wmma_f32_16x16x32_bf16(false, a, false, b,
                                                 (short)0, c, false, false);
}

static __device__ __forceinline__ v8f zero_v8f() {
  v8f z = {0.f, 0.f, 0.f, 0.f, 0.f, 0.f, 0.f, 0.f};
  return z;
}

// Async global -> LDS copy of 16 bytes (tracked by ASYNCcnt on CDNA5),
// with a synchronous VGPR-bounce fallback if the builtin is unavailable.
// Probe-derived signature: (v4i __device__* gsrc, v4i* lds, imm off, imm cpol)
static __device__ __forceinline__ void async_b128(const __bf16* g, __bf16* l) {
#ifdef HAS_ASYNC_LDS
  __builtin_amdgcn_global_load_async_to_lds_b128(
      (GLOBAL_AS v4i*)g, (LDS_AS v4i*)l, 0, 0);
#else
  *(v8bf*)l = *(const v8bf*)g;
#endif
}

static __device__ __forceinline__ void wait_async_zero() {
#ifdef HAS_ASYNC_LDS
#if __has_builtin(__builtin_amdgcn_s_wait_asynccnt)
  __builtin_amdgcn_s_wait_asynccnt(0);
#else
  asm volatile("s_wait_asynccnt 0" ::: "memory");
#endif
#endif
}

// ---------------------------------------------------------------------------
// Weight transpose + fp32 -> bf16 cast:  dst[c*rows + r] = src[r*cols + c]
// ---------------------------------------------------------------------------
__global__ void transpose_cast_kernel(const float* __restrict__ src,
                                      __bf16* __restrict__ dst,
                                      int rows, int cols) {
  int total = rows * cols;
  for (int i = blockIdx.x * blockDim.x + threadIdx.x; i < total;
       i += gridDim.x * blockDim.x) {
    int r = i / cols, c = i - r * cols;
    dst[(size_t)c * rows + r] = (__bf16)src[i];
  }
}

// ---------------------------------------------------------------------------
// LayerNorm1: wave-per-row (8 floats/lane), write bf16
// ---------------------------------------------------------------------------
__global__ __launch_bounds__(256) void ln1_kernel(
    const float* __restrict__ x, const float* __restrict__ g,
    const float* __restrict__ b, __bf16* __restrict__ out) {
  const int wave = threadIdx.x >> 5, lane = threadIdx.x & 31;
  const int row = blockIdx.x * 8 + wave;
  const float* xr = x + (size_t)row * C_DIM + lane * 8;
  float v[8]; float s = 0.f, s2 = 0.f;
#pragma unroll
  for (int i = 0; i < 8; ++i) { v[i] = xr[i]; s += v[i]; s2 += v[i] * v[i]; }
#pragma unroll
  for (int m = 16; m >= 1; m >>= 1) {
    s  += __shfl_xor(s,  m, 32);
    s2 += __shfl_xor(s2, m, 32);
  }
  const float mu = s * (1.f / C_DIM);
  const float ri = rsqrtf(s2 * (1.f / C_DIM) - mu * mu + EPSF);
  __bf16* orow = out + (size_t)row * C_DIM + lane * 8;
  const float* gp = g + lane * 8;
  const float* bp = b + lane * 8;
#pragma unroll
  for (int i = 0; i < 8; ++i)
    orow[i] = (__bf16)((v[i] - mu) * ri * gp[i] + bp[i]);
}

// ---------------------------------------------------------------------------
// WMMA GEMM v2: out[R x N] = A[R x K] @ Bt[N x K]^T (+bias)(+relu)(+resid)
// block = 256 thr = 8 waves covering a 32-row x 256-col output macro-tile.
// Wave (w>>2) picks the 16-row strip, (w&3) picks a 64-col strip = 4 tiles,
// so each A fragment is reused for 4 WMMAs.  The 256x32 Bt panel for each
// K-step is staged into LDS with double-buffered ASYNC loads: panel k+1 is
// in flight (ASYNCcnt) while WMMAs consume panel k from LDS.
// grid = (Ncols/256, R/32)
// ---------------------------------------------------------------------------
template <int OUT_BF16, int HAS_BIAS, int HAS_RELU, int HAS_RESID>
__global__ __launch_bounds__(256) void gemm_wmma_kernel(
    const __bf16* __restrict__ A, const __bf16* __restrict__ Bt,
    int K, int Ncols, float* __restrict__ outF, __bf16* __restrict__ outB,
    const float* __restrict__ bias, const float* __restrict__ resid) {
  __shared__ __align__(16) __bf16 bpanel[2][256 * 32];  // 2 x 16KB

  const int tid  = threadIdx.x;
  const int lane = tid & 31;
  const int wave = tid >> 5;
  const int l15  = lane & 15;
  const int koff = (lane >= 16) ? 8 : 0;
  const int hi8  = (lane >= 16) ? 8 : 0;

  const int colbase = blockIdx.x * 256;
  const int wrow    = wave >> 2;            // 0..1: 16-row strip
  const int wcol4   = wave & 3;             // 0..3: 64-col strip
  const int lrow    = blockIdx.y * 32 + wrow * 16 + l15;

  const __bf16* Arow = A + (size_t)lrow * K + koff;
  const __bf16* Brow = Bt + (size_t)(colbase + tid) * K;  // staging row

  const int nk = K >> 5;

  // Prologue: stage panel 0
#pragma unroll
  for (int j = 0; j < 4; ++j)
    async_b128(Brow + j * 8, &bpanel[0][tid * 32 + j * 8]);

  v8f acc[4] = {zero_v8f(), zero_v8f(), zero_v8f(), zero_v8f()};

  for (int kt = 0; kt < nk; ++kt) {
    const int k0 = kt * 32;
    wait_async_zero();
    __syncthreads();                 // panel (kt&1) visible to whole block
    if (kt + 1 < nk) {
      const __bf16* gsrc = Brow + k0 + 32;
      __bf16* ldst = &bpanel[(kt + 1) & 1][tid * 32];
#pragma unroll
      for (int j = 0; j < 4; ++j) async_b128(gsrc + j * 8, ldst + j * 8);
    }
    const __bf16* pan = &bpanel[kt & 1][0];
    v16bf a = load_frag16(Arow + k0);
#pragma unroll
    for (int c = 0; c < 4; ++c) {
      v16bf b = load_frag16(pan + (wcol4 * 64 + c * 16 + l15) * 32 + koff);
      acc[c] = wmma_bf16(a, b, acc[c]);
    }
    // No trailing barrier needed: next overwrite of this buffer is issued
    // only after the next top-of-loop barrier.
  }

  const int rbase = blockIdx.y * 32 + wrow * 16 + hi8;
#pragma unroll
  for (int c = 0; c < 4; ++c) {
    const int lcol = colbase + wcol4 * 64 + c * 16 + l15;
#pragma unroll
    for (int r = 0; r < 8; ++r) {
      const size_t o = (size_t)(rbase + r) * Ncols + lcol;
      float v = acc[c][r];
      if (HAS_BIAS)  v += bias[lcol];
      if (HAS_RELU)  v = fmaxf(v, 0.f);
      if (HAS_RESID) v += resid[o];
      if (OUT_BF16)  outB[o] = (__bf16)v;
      else           outF[o] = v;
    }
  }
}

// ---------------------------------------------------------------------------
// Windowed attention: one block (512 thr = 16 waves) per (window m, head h).
// Each wave owns a 16-query strip: scores (16x256) via 16 WMMAs (K=32 = d),
// fp32 masked softmax via half-wave shuffles, attn@V via WMMA with a per-wave
// 16x32 LDS transpose chunk and LDS-resident V^T.
// ---------------------------------------------------------------------------
__global__ __launch_bounds__(512) void attn_kernel(
    const __bf16* __restrict__ qkv, __bf16* __restrict__ attn_out) {
  __shared__ __align__(16) __bf16 vTs[D_HEAD * WIN_];      // V^T [32][256]
  __shared__ __align__(16) __bf16 chunkS[16][16 * 32];     // per-wave A-stage

  const int m = blockIdx.x >> 3;
  const int h = blockIdx.x & 7;
  const int start = m * STRIDE_;
  const int tid = threadIdx.x;
  const int lane = tid & 31;
  const int wave = tid >> 5;             // 0..15 = query strip
  const int l15 = lane & 15;
  const int koff = (lane >= 16) ? 8 : 0;
  const int hi8 = (lane >= 16) ? 8 : 0;

  // Stage V^T into LDS: vTs[dim][key]
  for (int i = tid; i < D_HEAD * WIN_; i += 512) {
    const int dd = i >> 8, kk = i & 255;
    int n = start + kk; if (n > N_PTS - 1) n = N_PTS - 1;
    vTs[i] = qkv[(size_t)n * C3 + 2 * C_DIM + h * D_HEAD + dd];
  }
  __syncthreads();

  // Q fragment for this strip (row = strip*16 + l15, K = d = 32)
  int qn = start + wave * 16 + l15; if (qn > N_PTS - 1) qn = N_PTS - 1;
  const v16bf aq = load_frag16(qkv + (size_t)qn * C3 + h * D_HEAD + koff);

  // Scores: 16 column tiles of 16x16
  v8f s[16];
#pragma unroll
  for (int ct = 0; ct < 16; ++ct) {
    int kn = start + ct * 16 + l15; if (kn > N_PTS - 1) kn = N_PTS - 1;
    v16bf bk = load_frag16(qkv + (size_t)kn * C3 + C_DIM + h * D_HEAD + koff);
    s[ct] = wmma_bf16(aq, bk, zero_v8f());
  }

  // Masked softmax over the 256 key columns of each of this wave's 16 rows.
  // C-layout: s[ct][r] lives at (row = strip*16 + r + hi8, col = ct*16 + l15);
  // rows are constant within each 16-lane half => width-16 shuffle reductions.
  const float scale = 0.17677669529663687f;  // 1/sqrt(32)
  bool cvalid[16];
#pragma unroll
  for (int ct = 0; ct < 16; ++ct) cvalid[ct] = (start + ct * 16 + l15) < N_PTS;
#pragma unroll
  for (int r = 0; r < 8; ++r) {
    float mx = -1e30f;
#pragma unroll
    for (int ct = 0; ct < 16; ++ct) {
      float v = cvalid[ct] ? s[ct][r] * scale : -1e30f;
      s[ct][r] = v;
      mx = fmaxf(mx, v);
    }
#pragma unroll
    for (int msk = 8; msk >= 1; msk >>= 1)
      mx = fmaxf(mx, __shfl_xor(mx, msk, 16));
    float sum = 0.f;
#pragma unroll
    for (int ct = 0; ct < 16; ++ct) {
      float e = __expf(s[ct][r] - mx);
      s[ct][r] = e;
      sum += e;
    }
#pragma unroll
    for (int msk = 8; msk >= 1; msk >>= 1) sum += __shfl_xor(sum, msk, 16);
    const float inv = 1.f / sum;
#pragma unroll
    for (int ct = 0; ct < 16; ++ct) s[ct][r] *= inv;
  }

  // attn @ V : out 16x32 per wave.  A-frags come from the per-wave LDS chunk
  // (C-layout -> A-layout transpose); Bt = V^T rows are contiguous in LDS.
  v8f o0 = zero_v8f(), o1 = zero_v8f();
  __bf16* chk = &chunkS[wave][0];
#pragma unroll
  for (int kt = 0; kt < 8; ++kt) {
    const int k0 = kt * 32;
#pragma unroll
    for (int r = 0; r < 8; ++r) {
      const int rr = r + hi8;
      chk[rr * 32 + l15]      = (__bf16)s[2 * kt][r];
      chk[rr * 32 + 16 + l15] = (__bf16)s[2 * kt + 1][r];
    }
    asm volatile("s_wait_dscnt 0" ::: "memory");  // in-wave LDS RAW
    v16bf a  = load_frag16(chk + l15 * 32 + koff);
    v16bf b0 = load_frag16(&vTs[l15 * WIN_ + k0 + koff]);
    v16bf b1 = load_frag16(&vTs[(16 + l15) * WIN_ + k0 + koff]);
    o0 = wmma_bf16(a, b0, o0);
    o1 = wmma_bf16(a, b1, o1);
  }

  // Store: attn_out[(m*WIN + wrow)*C + h*32 + dim] (bf16, feeds proj GEMM)
#pragma unroll
  for (int r = 0; r < 8; ++r) {
    const int wrow = wave * 16 + r + hi8;
    const size_t base = ((size_t)(m * WIN_ + wrow)) * C_DIM + h * D_HEAD;
    attn_out[base + l15]      = (__bf16)o0[r];
    attn_out[base + 16 + l15] = (__bf16)o1[r];
  }
}

// ---------------------------------------------------------------------------
// Overlap-gather (deterministic, atomic-free) + residual + LayerNorm2.
// Point n lives in window m0=n>>7 at row n&127, and (if m0>=1) window m0-1 at
// row (n&127)+128; both rows are always unmasked.  counts = 1 or 2.
// ---------------------------------------------------------------------------
__global__ __launch_bounds__(256) void residual_ln2_kernel(
    const float* __restrict__ x, const float* __restrict__ proj,
    const float* __restrict__ g, const float* __restrict__ b,
    float* __restrict__ xout, __bf16* __restrict__ lnout) {
  const int wave = threadIdx.x >> 5, lane = threadIdx.x & 31;
  const int row = blockIdx.x * 8 + wave;
  const int m0 = row >> 7, w0 = row & 127;
  const size_t r0 = ((size_t)(m0 * WIN_ + w0)) * C_DIM;
  const bool two = (m0 >= 1);
  const size_t r1 = two ? ((size_t)((m0 - 1) * WIN_ + w0 + 128)) * C_DIM : 0;
  const float inv = two ? 0.5f : 1.f;

  float v[8]; float s = 0.f, s2 = 0.f;
  const int cb = lane * 8;
#pragma unroll
  for (int i = 0; i < 8; ++i) {
    float acc = proj[r0 + cb + i];
    if (two) acc += proj[r1 + cb + i];
    float xo = x[(size_t)row * C_DIM + cb + i] + acc * inv;
    xout[(size_t)row * C_DIM + cb + i] = xo;
    v[i] = xo; s += xo; s2 += xo * xo;
  }
#pragma unroll
  for (int m = 16; m >= 1; m >>= 1) {
    s  += __shfl_xor(s,  m, 32);
    s2 += __shfl_xor(s2, m, 32);
  }
  const float mu = s * (1.f / C_DIM);
  const float ri = rsqrtf(s2 * (1.f / C_DIM) - mu * mu + EPSF);
  __bf16* orow = lnout + (size_t)row * C_DIM + cb;
#pragma unroll
  for (int i = 0; i < 8; ++i)
    orow[i] = (__bf16)((v[i] - mu) * ri * g[cb + i] + b[cb + i]);
}

// ---------------------------------------------------------------------------
// BatchNorm column stats over N rows (block per feature), then apply+relu.
// ---------------------------------------------------------------------------
__global__ __launch_bounds__(256) void bn_stats_kernel(
    const float* __restrict__ h, float* __restrict__ mu,
    float* __restrict__ ri) {
  __shared__ float sb[256], sb2[256];
  const int c = blockIdx.x;
  float s = 0.f, s2 = 0.f;
  for (int r = threadIdx.x; r < N_PTS; r += 256) {
    float v = h[(size_t)r * C2 + c];
    s += v; s2 += v * v;
  }
  sb[threadIdx.x] = s; sb2[threadIdx.x] = s2;
  __syncthreads();
  for (int st = 128; st > 0; st >>= 1) {
    if (threadIdx.x < st) {
      sb[threadIdx.x]  += sb[threadIdx.x + st];
      sb2[threadIdx.x] += sb2[threadIdx.x + st];
    }
    __syncthreads();
  }
  if (threadIdx.x == 0) {
    const float m = sb[0] * (1.f / N_PTS);
    mu[c] = m;
    ri[c] = rsqrtf(sb2[0] * (1.f / N_PTS) - m * m + EPSF);
  }
}

__global__ void bn_relu_kernel(const float* __restrict__ h,
                               const float* __restrict__ mu,
                               const float* __restrict__ ri,
                               const float* __restrict__ g,
                               const float* __restrict__ b,
                               __bf16* __restrict__ out) {
  const size_t total = (size_t)N_PTS * C2;
  for (size_t i = blockIdx.x * blockDim.x + threadIdx.x; i < total;
       i += (size_t)gridDim.x * blockDim.x) {
    const int c = (int)(i & (C2 - 1));
    float v = (h[i] - mu[c]) * ri[c] * g[c] + b[c];
    out[i] = (__bf16)fmaxf(v, 0.f);
  }
}

// ---------------------------------------------------------------------------
// Host side
// ---------------------------------------------------------------------------
extern "C" void kernel_launch(void* const* d_in, const int* in_sizes, int n_in,
                              void* d_out, int out_size, void* d_ws,
                              size_t ws_size, hipStream_t stream) {
  const float* x        = (const float*)d_in[0];
  // d_in[1] win_idx, d_in[2] win_mask: recomputed analytically on device
  const float* gamma1   = (const float*)d_in[3];
  const float* beta1    = (const float*)d_in[4];
  const float* w_qkv    = (const float*)d_in[5];
  const float* w_proj   = (const float*)d_in[6];
  const float* b_proj   = (const float*)d_in[7];
  const float* gamma2   = (const float*)d_in[8];
  const float* beta2    = (const float*)d_in[9];
  const float* w1       = (const float*)d_in[10];
  const float* b1       = (const float*)d_in[11];
  const float* bn_gamma = (const float*)d_in[12];
  const float* bn_beta  = (const float*)d_in[13];
  const float* w2       = (const float*)d_in[14];
  const float* b2       = (const float*)d_in[15];
  float* out = (float*)d_out;

  // Workspace carve-up (256B aligned)
  char* ws = (char*)d_ws;
  size_t off = 0;
  auto alloc = [&](size_t bytes) {
    void* p = ws + off;
    off += (bytes + 255) & ~(size_t)255;
    return p;
  };
  __bf16* xln    = (__bf16*)alloc((size_t)N_PTS * C_DIM * 2);
  __bf16* wqkvT  = (__bf16*)alloc((size_t)C3 * C_DIM * 2);
  __bf16* wprojT = (__bf16*)alloc((size_t)C_DIM * C_DIM * 2);
  __bf16* w1T    = (__bf16*)alloc((size_t)C2 * C_DIM * 2);
  __bf16* w2T    = (__bf16*)alloc((size_t)C_DIM * C2 * 2);
  __bf16* qkv    = (__bf16*)alloc((size_t)N_PTS * C3 * 2);
  __bf16* attnO  = (__bf16*)alloc((size_t)M_WIN * WIN_ * C_DIM * 2);
  float*  proj   = (float*)alloc((size_t)M_WIN * WIN_ * C_DIM * 4);
  float*  xoutB  = (float*)alloc((size_t)N_PTS * C_DIM * 4);
  __bf16* ln2    = (__bf16*)alloc((size_t)N_PTS * C_DIM * 2);
  float*  hbuf   = (float*)alloc((size_t)N_PTS * C2 * 4);
  __bf16* hrelu  = (__bf16*)alloc((size_t)N_PTS * C2 * 2);
  float*  bnmu   = (float*)alloc(C2 * 4);
  float*  bnri   = (float*)alloc(C2 * 4);
  (void)ws_size; (void)in_sizes; (void)n_in; (void)out_size;

  // 1) Weights -> bf16, transposed to [N x K]
  transpose_cast_kernel<<<512, 256, 0, stream>>>(w_qkv,  wqkvT,  C_DIM, C3);
  transpose_cast_kernel<<<256, 256, 0, stream>>>(w_proj, wprojT, C_DIM, C_DIM);
  transpose_cast_kernel<<<512, 256, 0, stream>>>(w1,     w1T,    C_DIM, C2);
  transpose_cast_kernel<<<512, 256, 0, stream>>>(w2,     w2T,    C2,    C_DIM);

  // 2) LayerNorm1 -> bf16
  ln1_kernel<<<N_PTS / 8, 256, 0, stream>>>(x, gamma1, beta1, xln);

  // 3) QKV = xln @ w_qkv   (32768 x 256 @ 256 x 768, bf16 out)
  gemm_wmma_kernel<1, 0, 0, 0><<<dim3(C3 / 256, N_PTS / 32), 256, 0, stream>>>(
      xln, wqkvT, C_DIM, C3, nullptr, qkv, nullptr, nullptr);

  // 4) Windowed attention (block per (m,h))
  attn_kernel<<<M_WIN * H_HEADS, 512, 0, stream>>>(qkv, attnO);

  // 5) proj = attnO @ w_proj + b_proj  (65536 x 256 @ 256 x 256, f32 out)
  gemm_wmma_kernel<0, 1, 0, 0>
      <<<dim3(C_DIM / 256, (M_WIN * WIN_) / 32), 256, 0, stream>>>(
          attnO, wprojT, C_DIM, C_DIM, proj, nullptr, b_proj, nullptr);

  // 6) overlap-gather + residual + LayerNorm2
  residual_ln2_kernel<<<N_PTS / 8, 256, 0, stream>>>(x, proj, gamma2, beta2,
                                                     xoutB, ln2);

  // 7) h = ln2 @ w1 + b1  (32768 x 256 @ 256 x 512, f32 out)
  gemm_wmma_kernel<0, 1, 0, 0><<<dim3(C2 / 256, N_PTS / 32), 256, 0, stream>>>(
      ln2, w1T, C_DIM, C2, hbuf, nullptr, b1, nullptr);

  // 8) BatchNorm stats + apply + relu -> bf16
  bn_stats_kernel<<<C2, 256, 0, stream>>>(hbuf, bnmu, bnri);
  bn_relu_kernel<<<8192, 256, 0, stream>>>(hbuf, bnmu, bnri, bn_gamma, bn_beta,
                                           hrelu);

  // 9) out = hrelu @ w2 + b2 + xout  (32768 x 512 @ 512 x 256, f32 out)
  gemm_wmma_kernel<0, 1, 0, 1><<<dim3(C_DIM / 256, N_PTS / 32), 256, 0, stream>>>(
      hrelu, w2T, C2, C_DIM, out, nullptr, b2, xoutB);
}
